// GroupedQueryAttentionWithALiBi_31112743092234
// MI455X (gfx1250) — matrix operand order repaired and checked by reading, not compiled, not run on hardware
//
#include <hip/hip_runtime.h>
#include <hip/hip_bf16.h>
#include <stdint.h>

#define NUM_Q_HEADS 32
#define NUM_KV_HEADS 8
#define HEAD_DIM 64
#define D_MODEL 2048
#define SEQ 2048
#define KV_DIM (HEAD_DIM * NUM_KV_HEADS) // 512

typedef __attribute__((ext_vector_type(16))) __bf16 v16bf;
typedef __attribute__((ext_vector_type(8)))  float  v8f;

struct alignas(16) U128 { unsigned x, y, z, w; };

union BFrag { v16bf v; U128 h[2]; unsigned u[8]; };

__device__ __host__ inline unsigned short f2bf(float f) {
  union { float f; unsigned u; } c; c.f = f;
  unsigned r = c.u + 0x7FFFu + ((c.u >> 16) & 1u); // round-to-nearest-even
  return (unsigned short)(r >> 16);
}

__device__ inline v8f bwmma(const BFrag& a, const BFrag& b, v8f c) {
  return __builtin_amdgcn_wmma_f32_16x16x32_bf16(
      false, a.v, false, b.v, (short)0, c, false, false);
}

// ---------------- f32 -> bf16 conversion ----------------
__global__ void cvt_f32_to_bf16(const float* __restrict__ s,
                                unsigned short* __restrict__ d, int n) {
  int i = blockIdx.x * blockDim.x + threadIdx.x;
  if (i < n) d[i] = f2bf(s[i]);
}

// ---------------- GEMM: C = A(MxK) * B(NxK)^T + bias ----------------
// A, B bf16 row-major. Each wave computes a 32(M) x 64(N) tile:
// two M-tiles share the four B fragments -> 12 b128 loads per 8 WMMAs.
// MODE 0: f32 out row-major (MxN).
// MODE 1: bf16 out row-major, value scaled by `scale` after bias.
// MODE 2: bf16 out transposed (NxM) -- used to produce V^T.
template <int MODE>
__global__ __launch_bounds__(256) void gemm_bf16_wmma(
    const unsigned short* __restrict__ A, const unsigned short* __restrict__ Bw,
    const float* __restrict__ bias, void* __restrict__ outp,
    int M, int N, int K, float scale)
{
  const int lane  = threadIdx.x & 31;
  const int wave  = threadIdx.x >> 5;
  const int l16   = lane & 15;
  const int hi    = (lane >> 4) & 1;
  const int mbase = blockIdx.y << 5;              // 32 rows per block
  const int nbase = (blockIdx.x << 9) + (wave << 6);
  if (nbase >= N) return; // wave-uniform

  v8f acc[2][4] = {};

  // A fragment row pointers for the two 16-row M-tiles
  const unsigned short* arow0 = A + (size_t)(mbase + l16) * K + hi * 8;
  const unsigned short* arow1 = arow0 + (size_t)16 * K;

  #pragma unroll 1
  for (int k0 = 0; k0 < K; k0 += 32) {
    // prime L2 for A eight K-blocks (256 elements) ahead (speculative, no fault)
    __builtin_prefetch(arow0 + k0 + 256, 0, 1);
    __builtin_prefetch(arow1 + k0 + 256, 0, 1);

    BFrag a0, a1;
    a0.h[0] = *(const U128*)(arow0 + k0);        // K = k0 + (hi?8:0) + 0..7
    a0.h[1] = *(const U128*)(arow0 + k0 + 16);   // K = k0+16 + (hi?8:0) + 0..7
    a1.h[0] = *(const U128*)(arow1 + k0);
    a1.h[1] = *(const U128*)(arow1 + k0 + 16);
    #pragma unroll
    for (int t = 0; t < 4; ++t) {
      const unsigned short* bp =
          Bw + (size_t)(nbase + t * 16 + l16) * K + k0 + hi * 8;
      BFrag b;
      b.h[0] = *(const U128*)(bp);               // K = k0 + (hi?8:0) + 0..7
      b.h[1] = *(const U128*)(bp + 16);          // K = k0+16 + (hi?8:0) + 0..7
      acc[0][t] = bwmma(a0, b, acc[0][t]);
      acc[1][t] = bwmma(a1, b, acc[1][t]);
    }
  }

  #pragma unroll
  for (int mt = 0; mt < 2; ++mt) {
    const int mrow = mbase + mt * 16 + hi * 8;
    #pragma unroll
    for (int t = 0; t < 4; ++t) {
      const int n = nbase + t * 16 + l16;
      const float bv = bias ? bias[n] : 0.0f;
      if (MODE == 0) {
        float* o = (float*)outp;
        #pragma unroll
        for (int r = 0; r < 8; ++r)
          o[(size_t)(mrow + r) * N + n] = acc[mt][t][r] + bv;
      } else if (MODE == 1) {
        unsigned short* o = (unsigned short*)outp;
        #pragma unroll
        for (int r = 0; r < 8; ++r)
          o[(size_t)(mrow + r) * N + n] = f2bf((acc[mt][t][r] + bv) * scale);
      } else {
        unsigned short* o = (unsigned short*)outp;
        U128 pk;
        pk.x = (unsigned)f2bf(acc[mt][t][0] + bv) | ((unsigned)f2bf(acc[mt][t][1] + bv) << 16);
        pk.y = (unsigned)f2bf(acc[mt][t][2] + bv) | ((unsigned)f2bf(acc[mt][t][3] + bv) << 16);
        pk.z = (unsigned)f2bf(acc[mt][t][4] + bv) | ((unsigned)f2bf(acc[mt][t][5] + bv) << 16);
        pk.w = (unsigned)f2bf(acc[mt][t][6] + bv) | ((unsigned)f2bf(acc[mt][t][7] + bv) << 16);
        *(U128*)(o + (size_t)n * M + mrow) = pk; // transposed, 8 contiguous tokens
      }
    }
  }
}

// ---------------- Flash-style GQA attention with ALiBi ----------------
// One wave per (q-head, 16-query block). Q: (seq, 2048) bf16 (pre-scaled by 1/8),
// K: (seq, 512) bf16, Vt: (512, seq) bf16 (transposed). Ctx out: (seq, 2048) bf16.
__global__ __launch_bounds__(128) void attn_gqa_alibi(
    const unsigned short* __restrict__ Qb, const unsigned short* __restrict__ Kb,
    const unsigned short* __restrict__ Vt, unsigned short* __restrict__ Ctx)
{
  __shared__ __align__(16) unsigned short plds[4 * 16 * 40]; // per-wave 16x32 P tile, stride 40

  const int lane = threadIdx.x & 31;
  const int wave = threadIdx.x >> 5;
  const int l16  = lane & 15;
  const int hi   = (lane >> 4) & 1;

  const int w    = blockIdx.x * 4 + wave;
  const int qh   = w >> 7;     // 0..31  (consecutive waves share K/V of one head)
  const int qblk = w & 127;    // 0..127
  const int i0   = qblk << 4;
  const int h    = qh >> 2;    // kv head = qh // g
  const int gi   = qh & 3;     // group index = qh % g
  const float slope = exp2f(-0.25f * (float)gi);

  unsigned short* P = plds + wave * (16 * 40);

  // Q A-fragments for K-dims [0,32) and [32,64); Q already scaled by 1/sqrt(d)
  const unsigned short* qrow =
      Qb + (size_t)(i0 + l16) * D_MODEL + qh * HEAD_DIM + hi * 8;
  BFrag qa0, qa1;
  qa0.h[0] = *(const U128*)(qrow);
  qa0.h[1] = *(const U128*)(qrow + 16);
  qa1.h[0] = *(const U128*)(qrow + 32);
  qa1.h[1] = *(const U128*)(qrow + 48);

  float mx[8], ls[8], sc[8];
  #pragma unroll
  for (int r = 0; r < 8; ++r) { mx[r] = -3.0e38f; ls[r] = 0.0f; }
  v8f o[4] = {};

  const int irow = i0 + hi * 8;

  #pragma unroll 1
  for (int jb = 0; jb < SEQ; jb += 32) {
    // --- scores: two 16x16 tiles over keys [jb, jb+16) and [jb+16, jb+32) ---
    const unsigned short* k0p =
        Kb + (size_t)(jb + l16) * KV_DIM + h * HEAD_DIM + hi * 8;
    const unsigned short* k1p = k0p + (size_t)16 * KV_DIM;
    BFrag kb;
    v8f s0 = {}, s1 = {};
    kb.h[0] = *(const U128*)(k0p);      kb.h[1] = *(const U128*)(k0p + 16);
    s0 = bwmma(qa0, kb, s0);
    kb.h[0] = *(const U128*)(k0p + 32); kb.h[1] = *(const U128*)(k0p + 48);
    s0 = bwmma(qa1, kb, s0);
    kb.h[0] = *(const U128*)(k1p);      kb.h[1] = *(const U128*)(k1p + 16);
    s1 = bwmma(qa0, kb, s1);
    kb.h[0] = *(const U128*)(k1p + 32); kb.h[1] = *(const U128*)(k1p + 48);
    s1 = bwmma(qa1, kb, s1);

    // --- ALiBi bias + online softmax (rows live across 16-lane groups) ---
    const float j0 = (float)(jb + l16);
    #pragma unroll
    for (int r = 0; r < 8; ++r) {
      const float fi = (float)(irow + r);
      float p0 = s0[r] - slope * fabsf(j0 - fi);
      float p1 = s1[r] - slope * fabsf(j0 + 16.0f - fi);
      float cm = fmaxf(p0, p1);
      cm = fmaxf(cm, __shfl_xor(cm, 1, 32));
      cm = fmaxf(cm, __shfl_xor(cm, 2, 32));
      cm = fmaxf(cm, __shfl_xor(cm, 4, 32));
      cm = fmaxf(cm, __shfl_xor(cm, 8, 32));
      const float nm   = fmaxf(mx[r], cm);
      const float corr = __expf(mx[r] - nm);
      const float e0   = __expf(p0 - nm);
      const float e1   = __expf(p1 - nm);
      float ps = e0 + e1;
      ps += __shfl_xor(ps, 1, 32);
      ps += __shfl_xor(ps, 2, 32);
      ps += __shfl_xor(ps, 4, 32);
      ps += __shfl_xor(ps, 8, 32);
      ls[r] = ls[r] * corr + ps;
      mx[r] = nm;
      sc[r] = corr;
      const int m = hi * 8 + r;                 // C-layout row
      P[m * 40 + l16]      = f2bf(e0);          // transpose via LDS
      P[m * 40 + 16 + l16] = f2bf(e1);
    }
    #pragma unroll
    for (int r = 0; r < 8; ++r) {
      o[0][r] *= sc[r]; o[1][r] *= sc[r]; o[2][r] *= sc[r]; o[3][r] *= sc[r];
    }

    // --- P (16x32) as A-fragment from LDS; PV over V^T (contiguous loads) ---
    const unsigned short* prow = P + l16 * 40 + hi * 8;
    BFrag pa;
    pa.h[0] = *(const U128*)(prow);
    pa.h[1] = *(const U128*)(prow + 16);
    #pragma unroll
    for (int t = 0; t < 4; ++t) {
      const unsigned short* vp =
          Vt + (size_t)(h * HEAD_DIM + t * 16 + l16) * SEQ + jb + hi * 8;
      BFrag vb;
      vb.h[0] = *(const U128*)(vp);
      vb.h[1] = *(const U128*)(vp + 16);
      o[t] = bwmma(pa, vb, o[t]);
    }
  }

  // --- normalize and store context (natural qh*64+d column ordering) ---
  #pragma unroll
  for (int t = 0; t < 4; ++t) {
    const int col = qh * HEAD_DIM + t * 16 + l16;
    #pragma unroll
    for (int r = 0; r < 8; ++r) {
      Ctx[(size_t)(irow + r) * D_MODEL + col] = f2bf(o[t][r] / ls[r]);
    }
  }
}

extern "C" void kernel_launch(void* const* d_in, const int* in_sizes, int n_in,
                              void* d_out, int out_size, void* d_ws, size_t ws_size,
                              hipStream_t stream) {
  (void)in_sizes; (void)n_in; (void)out_size; (void)ws_size;
  const float* hs = (const float*)d_in[0];
  const float* Wq = (const float*)d_in[1];
  const float* bq = (const float*)d_in[2];
  const float* Wk = (const float*)d_in[3];
  const float* bk = (const float*)d_in[4];
  const float* Wv = (const float*)d_in[5];
  const float* bv = (const float*)d_in[6];
  const float* Wo = (const float*)d_in[7];
  float* out = (float*)d_out;

  char* ws = (char*)d_ws;
  size_t off = 0;
  auto wsalloc = [&](size_t bytes) -> void* {
    void* p = ws + off;
    off += (bytes + 255) & ~(size_t)255;
    return p;
  };
  unsigned short* Xb   = (unsigned short*)wsalloc((size_t)SEQ * D_MODEL * 2);
  unsigned short* Wqb  = (unsigned short*)wsalloc((size_t)D_MODEL * D_MODEL * 2);
  unsigned short* Wkb  = (unsigned short*)wsalloc((size_t)KV_DIM * D_MODEL * 2);
  unsigned short* Wvb  = (unsigned short*)wsalloc((size_t)KV_DIM * D_MODEL * 2);
  unsigned short* Wob  = (unsigned short*)wsalloc((size_t)D_MODEL * D_MODEL * 2);
  unsigned short* Qbf  = (unsigned short*)wsalloc((size_t)SEQ * D_MODEL * 2);
  unsigned short* Kbf  = (unsigned short*)wsalloc((size_t)SEQ * KV_DIM * 2);
  unsigned short* Vtb  = (unsigned short*)wsalloc((size_t)KV_DIM * SEQ * 2);
  unsigned short* Ctxb = (unsigned short*)wsalloc((size_t)SEQ * D_MODEL * 2);

  auto cvt = [&](const float* s, unsigned short* d, int n) {
    cvt_f32_to_bf16<<<dim3((n + 255) / 256), 256, 0, stream>>>(s, d, n);
  };
  cvt(hs, Xb,  SEQ * D_MODEL);
  cvt(Wq, Wqb, D_MODEL * D_MODEL);
  cvt(Wk, Wkb, KV_DIM * D_MODEL);
  cvt(Wv, Wvb, KV_DIM * D_MODEL);
  cvt(Wo, Wob, D_MODEL * D_MODEL);

  // Q = (X Wq^T + bq) * 1/sqrt(64), bf16 row-major
  gemm_bf16_wmma<1><<<dim3(D_MODEL / 512, SEQ / 32), 256, 0, stream>>>(
      Xb, Wqb, bq, Qbf, SEQ, D_MODEL, D_MODEL, 0.125f);
  // K = X Wk^T + bk, bf16 row-major
  gemm_bf16_wmma<1><<<dim3(1, SEQ / 32), 256, 0, stream>>>(
      Xb, Wkb, bk, Kbf, SEQ, KV_DIM, D_MODEL, 1.0f);
  // V^T = (X Wv^T + bv)^T, bf16 (512 x seq)
  gemm_bf16_wmma<2><<<dim3(1, SEQ / 32), 256, 0, stream>>>(
      Xb, Wvb, bv, Vtb, SEQ, KV_DIM, D_MODEL, 1.0f);

  // attention: 32 heads * 128 query blocks = 4096 waves, 4 waves/block
  attn_gqa_alibi<<<dim3((NUM_Q_HEADS * (SEQ / 16)) / 4), 128, 0, stream>>>(
      Qbf, Kbf, Vtb, Ctxb);

  // out = Ctx Wo^T, f32
  gemm_bf16_wmma<0><<<dim3(D_MODEL / 512, SEQ / 32), 256, 0, stream>>>(
      Ctxb, Wob, nullptr, out, SEQ, D_MODEL, D_MODEL, 1.0f);
}
